// MultibandFrameAttention_1279900254347
// MI455X (gfx1250) — compile-verified, matrix operand order checked by reading, not run on hardware
//
#include <hip/hip_runtime.h>
#include <hip/hip_bf16.h>
#include <math.h>

// ---------------------------------------------------------------------------
// MultibandFrameAttention for MI455X (gfx1250), wave32 + WMMA f16->f32,
// with async global->LDS tile staging (ASYNCcnt) and double buffering.
// ---------------------------------------------------------------------------

typedef __attribute__((ext_vector_type(16))) _Float16 v16h;
typedef __attribute__((ext_vector_type(8)))  float    v8f;
typedef __attribute__((ext_vector_type(4)))  int      i32x4;

#define DEVINL static __device__ __forceinline__

static constexpr int Wn   = 2048;   // frames
static constexpr int BINS = 1024;
static constexpr int H    = 16;     // bands
static constexpr int D    = 64;     // per-band dim
static constexpr int CROP = 2048;

// ---- async global->LDS (gfx1250): guarded so compile never regresses -------
#if defined(__has_builtin)
#  if __has_builtin(__builtin_amdgcn_global_load_async_to_lds_b128) && \
      __has_builtin(__builtin_amdgcn_s_wait_asynccnt)
#    define ASYNC_LDS 1
#  endif
#endif
#ifndef ASYNC_LDS
#  define ASYNC_LDS 0
#endif

DEVINL void cp16_async(_Float16* lds_dst, const _Float16* gsrc) {
#if ASYNC_LDS
  __builtin_amdgcn_global_load_async_to_lds_b128(
      (__attribute__((address_space(1))) i32x4*)(gsrc),
      (__attribute__((address_space(3))) i32x4*)(lds_dst),
      /*offset=*/0, /*cpol=*/0);
#else
  *reinterpret_cast<uint4*>(lds_dst) = *reinterpret_cast<const uint4*>(gsrc);
#endif
}

template <int N>
DEVINL void async_wait() {
#if ASYNC_LDS
  __builtin_amdgcn_s_wait_asynccnt(N);
#endif
}

// ---------------------------------------------------------------------------

union U32H2 { unsigned u; _Float16 h[2]; };

// Load one 16x32 f16 fragment (WMMA A layout; B uses the same layout on a
// transposed [N x K] buffer). Lane l holds row (l&15); element pair e=2r,2r+1
// maps to K = (e&8)*2 + ((l>>4)*8) + (e&7) -> contiguous dword loads.
DEVINL v16h lds_frag(const _Float16* lds, int row_off, int ld, int k_off) {
  const int lane = threadIdx.x & 31;
  const _Float16* p = lds + (row_off + (lane & 15)) * ld + k_off;
  const int kg = (lane >> 4) << 3;   // 0 or 8
  v16h f;
#pragma unroll
  for (int r = 0; r < 8; ++r) {
    const int k = ((r & 4) << 2) | kg | ((r & 3) << 1);
    U32H2 t;
    t.u = *reinterpret_cast<const unsigned*>(p + k);
    f[2 * r]     = t.h[0];
    f[2 * r + 1] = t.h[1];
  }
  return f;
}

DEVINL v8f wmma_f16(v16h a, v16h b, v8f c) {
  return __builtin_amdgcn_wmma_f32_16x16x32_f16(
      /*neg_a=*/false, a, /*neg_b=*/false, b,
      /*c_mod=*/(short)0, c, /*reuse_a=*/false, /*reuse_b=*/false);
}

DEVINL v8f v8f_zero() {
  v8f z;
#pragma unroll
  for (int i = 0; i < 8; ++i) z[i] = 0.0f;
  return z;
}

// ---------------------------------------------------------------------------
// Prep kernels
// ---------------------------------------------------------------------------

__global__ void cvt_f32_f16(const float* __restrict__ src,
                            _Float16* __restrict__ dst, int n) {
  int i = blockIdx.x * 256 + threadIdx.x;
  if (i < n) dst[i] = (_Float16)src[i];
}

// dst[c*R + r] = (f16) src[r*C + c]   (R x C  ->  C x R)
__global__ void transpose_f32_f16(const float* __restrict__ src,
                                  _Float16* __restrict__ dst, int R, int C) {
  int i = blockIdx.x * 256 + threadIdx.x;
  if (i >= R * C) return;
  int r = i / C, c = i % C;
  dst[(size_t)c * R + r] = (_Float16)src[(size_t)r * C + c];
}

// ---------------------------------------------------------------------------
// Generic WMMA GEMM: C[MxN] = A[MxK] * Bt[NxK]^T   (f16 in, f32 out)
// Macro tile 128x64, 8 waves (4x2), 32x32 per wave (2x2 frags), K step 32.
// Double-buffered LDS with async global->LDS DMA: tile k+1 streams in while
// tile k is consumed by WMMA.
// ---------------------------------------------------------------------------

__global__ void __launch_bounds__(256)
gemm_f16(const _Float16* __restrict__ A, const _Float16* __restrict__ Bt,
         float* __restrict__ C, int M, int N, int K) {
  __shared__ _Float16 sA[2][128 * 32];
  __shared__ _Float16 sB[2][64 * 32];

  const int tiles_n = N >> 6;
  const int m0 = (blockIdx.x / tiles_n) << 7;
  const int n0 = (blockIdx.x % tiles_n) << 6;
  const int t = threadIdx.x;
  const int lane = t & 31, wv = t >> 5;
  const int wm = (wv & 3) << 5;    // 0,32,64,96
  const int wn = (wv >> 2) << 5;   // 0,32

  v8f acc[2][2];
  acc[0][0] = v8f_zero(); acc[0][1] = v8f_zero();
  acc[1][0] = v8f_zero(); acc[1][1] = v8f_zero();

  const int arow = t >> 1, acol = (t & 1) << 4;   // A: 16 halves / thread
  const int brow = t >> 2, bcol = (t & 3) << 3;   // B: 8 halves / thread
  const _Float16* gA = A + (size_t)(m0 + arow) * K + acol;
  const _Float16* gB = Bt + (size_t)(n0 + brow) * K + bcol;

  // Stage one 128x32 A tile + 64x32 B tile into buffer `buf` (3 b128 / thread).
  auto stage = [&](int buf, int k0) {
    cp16_async(&sA[buf][arow * 32 + acol],     gA + k0);
    cp16_async(&sA[buf][arow * 32 + acol + 8], gA + k0 + 8);
    cp16_async(&sB[buf][brow * 32 + bcol],     gB + k0);
  };

  const int nk = K >> 5;
  stage(0, 0);

  for (int kt = 0; kt < nk; ++kt) {
    const int cur = kt & 1;
    if (kt + 1 < nk) {
      stage(cur ^ 1, (kt + 1) << 5);
      async_wait<3>();   // current tile's 3 DMAs done; next tile in flight
    } else {
      async_wait<0>();
    }
    __syncthreads();

    v16h af0 = lds_frag(sA[cur], wm,      32, 0);
    v16h af1 = lds_frag(sA[cur], wm + 16, 32, 0);
    v16h bf0 = lds_frag(sB[cur], wn,      32, 0);
    v16h bf1 = lds_frag(sB[cur], wn + 16, 32, 0);

    acc[0][0] = wmma_f16(af0, bf0, acc[0][0]);
    acc[0][1] = wmma_f16(af0, bf1, acc[0][1]);
    acc[1][0] = wmma_f16(af1, bf0, acc[1][0]);
    acc[1][1] = wmma_f16(af1, bf1, acc[1][1]);
    __syncthreads();   // all waves done reading buf[cur] before it's re-staged
  }

  const int crow = (lane >> 4) << 3;  // 0 or 8
  const int ccol = lane & 15;
#pragma unroll
  for (int fm = 0; fm < 2; ++fm)
#pragma unroll
    for (int fn = 0; fn < 2; ++fn)
#pragma unroll
      for (int r = 0; r < 8; ++r) {
        const int row = m0 + wm + fm * 16 + r + crow;
        const int col = n0 + wn + fn * 16 + ccol;
        C[(size_t)row * N + col] = acc[fm][fn][r];
      }
}

// ---------------------------------------------------------------------------
// Depthwise conv (K=3, pad=1) + band split.
// qh,kh : (H, W, D) f16 ; vT : (H, D, W) f16 (== (c, w) since h*64+d = c)
// ---------------------------------------------------------------------------

__global__ void __launch_bounds__(256)
conv_bands(const float* __restrict__ xq, const float* __restrict__ xk,
           const float* __restrict__ xv, const float* __restrict__ cq,
           const float* __restrict__ ck, const float* __restrict__ cv,
           _Float16* __restrict__ qh, _Float16* __restrict__ kh,
           _Float16* __restrict__ vT) {
  const int idx = blockIdx.x * 256 + threadIdx.x;  // over W*BINS
  const int c = idx & (BINS - 1);
  const int w = idx >> 10;
  const float xm_q = (w > 0)       ? xq[(size_t)(w - 1) * BINS + c] : 0.f;
  const float x0_q =                 xq[(size_t)w * BINS + c];
  const float xp_q = (w < Wn - 1)  ? xq[(size_t)(w + 1) * BINS + c] : 0.f;
  const float xm_k = (w > 0)       ? xk[(size_t)(w - 1) * BINS + c] : 0.f;
  const float x0_k =                 xk[(size_t)w * BINS + c];
  const float xp_k = (w < Wn - 1)  ? xk[(size_t)(w + 1) * BINS + c] : 0.f;
  const float xm_v = (w > 0)       ? xv[(size_t)(w - 1) * BINS + c] : 0.f;
  const float x0_v =                 xv[(size_t)w * BINS + c];
  const float xp_v = (w < Wn - 1)  ? xv[(size_t)(w + 1) * BINS + c] : 0.f;

  const float qv = xm_q * cq[c * 3] + x0_q * cq[c * 3 + 1] + xp_q * cq[c * 3 + 2];
  const float kv = xm_k * ck[c * 3] + x0_k * ck[c * 3 + 1] + xp_k * ck[c * 3 + 2];
  const float vv = xm_v * cv[c * 3] + x0_v * cv[c * 3 + 1] + xp_v * cv[c * 3 + 2];

  const int h = c >> 6, d = c & 63;
  const size_t band_idx = ((size_t)h * Wn + w) * D + d;
  qh[band_idx] = (_Float16)qv;
  kh[band_idx] = (_Float16)kv;
  vT[(size_t)c * Wn + w] = (_Float16)vv;  // h*64+d == c
}

// ---------------------------------------------------------------------------
// QK kernel: qk[h,i,j] = (q_i . k_j  +  rwT_i . q_j) / 32 + prev[h,i,j]
// The second GEMM realizes the pad/transpose/slice bias: p[i,j] = s[j,i].
// Macro tile 128(i) x 64(j); K = 64 fully resident in LDS, staged by async
// DMA (12 b128 per thread) -> single wait + barrier.
// ---------------------------------------------------------------------------

__global__ void __launch_bounds__(256)
qk_kernel(const _Float16* __restrict__ qh, const _Float16* __restrict__ kh,
          const _Float16* __restrict__ rwT, const float* __restrict__ prev,
          float* __restrict__ qk) {
  __shared__ _Float16 sQ[128 * 64];   // A1: q rows i
  __shared__ _Float16 sR[128 * 64];   // A2: relw^T rows i
  __shared__ _Float16 sK[64 * 64];    // B1: k rows j
  __shared__ _Float16 sQj[64 * 64];   // B2: q rows j

  const int n0 = blockIdx.x << 6;   // j tile (32 tiles)
  const int m0 = blockIdx.y << 7;   // i tile (16 tiles)
  const int h  = blockIdx.z;        // band  (16)
  const _Float16* qb = qh + (size_t)h * Wn * D;
  const _Float16* kb = kh + (size_t)h * Wn * D;

  const int t = threadIdx.x;
  {  // 128x64 tiles: 32 halves / thread (4 b128 each)
    const int row = t >> 1, col = (t & 1) << 5;
    const _Float16* g1 = qb  + (size_t)(m0 + row) * D + col;
    const _Float16* g2 = rwT + (size_t)(m0 + row) * D + col;
#pragma unroll
    for (int u = 0; u < 4; ++u) {
      cp16_async(sQ + row * 64 + col + 8 * u, g1 + 8 * u);
      cp16_async(sR + row * 64 + col + 8 * u, g2 + 8 * u);
    }
  }
  {  // 64x64 tiles: 16 halves / thread (2 b128 each)
    const int row = t >> 2, col = (t & 3) << 4;
    const _Float16* g1 = kb + (size_t)(n0 + row) * D + col;
    const _Float16* g2 = qb + (size_t)(n0 + row) * D + col;
#pragma unroll
    for (int u = 0; u < 2; ++u) {
      cp16_async(sK  + row * 64 + col + 8 * u, g1 + 8 * u);
      cp16_async(sQj + row * 64 + col + 8 * u, g2 + 8 * u);
    }
  }
  async_wait<0>();
  __syncthreads();

  const int lane = t & 31, wv = t >> 5;
  const int wm = (wv & 3) << 5, wn = (wv >> 2) << 5;

  v8f acc[2][2];
  acc[0][0] = v8f_zero(); acc[0][1] = v8f_zero();
  acc[1][0] = v8f_zero(); acc[1][1] = v8f_zero();

#pragma unroll
  for (int k0 = 0; k0 < 64; k0 += 32) {
    v16h a1_0 = lds_frag(sQ,  wm,      64, k0);
    v16h a1_1 = lds_frag(sQ,  wm + 16, 64, k0);
    v16h a2_0 = lds_frag(sR,  wm,      64, k0);
    v16h a2_1 = lds_frag(sR,  wm + 16, 64, k0);
    v16h b1_0 = lds_frag(sK,  wn,      64, k0);
    v16h b1_1 = lds_frag(sK,  wn + 16, 64, k0);
    v16h b2_0 = lds_frag(sQj, wn,      64, k0);
    v16h b2_1 = lds_frag(sQj, wn + 16, 64, k0);

    acc[0][0] = wmma_f16(a1_0, b1_0, acc[0][0]);
    acc[0][1] = wmma_f16(a1_0, b1_1, acc[0][1]);
    acc[1][0] = wmma_f16(a1_1, b1_0, acc[1][0]);
    acc[1][1] = wmma_f16(a1_1, b1_1, acc[1][1]);
    acc[0][0] = wmma_f16(a2_0, b2_0, acc[0][0]);
    acc[0][1] = wmma_f16(a2_0, b2_1, acc[0][1]);
    acc[1][0] = wmma_f16(a2_1, b2_0, acc[1][0]);
    acc[1][1] = wmma_f16(a2_1, b2_1, acc[1][1]);
  }

  const float inv_sqrt_c = 0.03125f;  // 1/sqrt(1024)
  const size_t base = (size_t)h * Wn * Wn;
  const int crow = (lane >> 4) << 3, ccol = lane & 15;
#pragma unroll
  for (int fm = 0; fm < 2; ++fm)
#pragma unroll
    for (int fn = 0; fn < 2; ++fn)
#pragma unroll
      for (int r = 0; r < 8; ++r) {
        const int row = m0 + wm + fm * 16 + r + crow;
        const int col = n0 + wn + fn * 16 + ccol;
        const size_t idx = base + (size_t)row * Wn + col;
        qk[idx] = acc[fm][fn][r] * inv_sqrt_c + prev[idx];
      }
}

// ---------------------------------------------------------------------------
// Row softmax stats: max and 1/sum(exp) per (h,i) row of qk.
// ---------------------------------------------------------------------------

__global__ void __launch_bounds__(256)
softmax_stats(const float* __restrict__ qk, float* __restrict__ rmax,
              float* __restrict__ rinv) {
  __shared__ float red[256];
  const int row = blockIdx.x;  // 0 .. H*W-1
  const float* p = qk + (size_t)row * Wn;
  const int t = threadIdx.x;

  float m = -INFINITY;
  for (int j = t; j < Wn; j += 256) m = fmaxf(m, p[j]);
  red[t] = m; __syncthreads();
  for (int s = 128; s > 0; s >>= 1) {
    if (t < s) red[t] = fmaxf(red[t], red[t + s]);
    __syncthreads();
  }
  m = red[0]; __syncthreads();

  float sum = 0.f;
  for (int j = t; j < Wn; j += 256) sum += __expf(p[j] - m);
  red[t] = sum; __syncthreads();
  for (int s = 128; s > 0; s >>= 1) {
    if (t < s) red[t] += red[t + s];
    __syncthreads();
  }
  if (t == 0) { rmax[row] = m; rinv[row] = 1.0f / red[0]; }
}

// ---------------------------------------------------------------------------
// AV kernel: o[h,i,d] = (1/sum_i) * sum_j exp(qk[h,i,j]-max_i) * v[h,j,d]
// Probabilities built on-the-fly in f16 LDS; V tile streamed by async DMA
// overlapping the exp-conversion VALU work.
// ---------------------------------------------------------------------------

__global__ void __launch_bounds__(256)
av_kernel(const float* __restrict__ qk, const _Float16* __restrict__ vT,
          const float* __restrict__ rmax, const float* __restrict__ rinv,
          _Float16* __restrict__ oh) {
  __shared__ _Float16 sP[128 * 32];
  __shared__ _Float16 sV[64 * 32];

  const int m0 = blockIdx.x << 7;  // i tile (16)
  const int h  = blockIdx.y;       // band (16)
  const float* qkb = qk + ((size_t)h * Wn + m0) * Wn;
  const _Float16* vb = vT + (size_t)h * D * Wn;
  const float* rmx = rmax + h * Wn + m0;
  const float* rin = rinv + h * Wn + m0;

  const int t = threadIdx.x, lane = t & 31, wv = t >> 5;
  const int wm = (wv & 3) << 5, wn = (wv >> 2) << 5;

  const int prow = t >> 1, pcol = (t & 1) << 4;  // 16 probs / thread
  const int vrow = t >> 2, vcol = (t & 3) << 3;  // 8 halves / thread
  const float pm = rmx[prow];

  v8f acc[2][2];
  acc[0][0] = v8f_zero(); acc[0][1] = v8f_zero();
  acc[1][0] = v8f_zero(); acc[1][1] = v8f_zero();

  for (int j0 = 0; j0 < Wn; j0 += 32) {
    // V tile DMA in flight while we compute probabilities.
    cp16_async(sV + vrow * 32 + vcol, vb + (size_t)vrow * Wn + j0 + vcol);
    {
      const float4* g = reinterpret_cast<const float4*>(
          qkb + (size_t)prow * Wn + j0 + pcol);
      _Float16* s = sP + prow * 32 + pcol;
#pragma unroll
      for (int u = 0; u < 4; ++u) {
        const float4 x = g[u];
        s[4 * u + 0] = (_Float16)__expf(x.x - pm);
        s[4 * u + 1] = (_Float16)__expf(x.y - pm);
        s[4 * u + 2] = (_Float16)__expf(x.z - pm);
        s[4 * u + 3] = (_Float16)__expf(x.w - pm);
      }
    }
    async_wait<0>();
    __syncthreads();

    v16h af0 = lds_frag(sP, wm,      32, 0);
    v16h af1 = lds_frag(sP, wm + 16, 32, 0);
    v16h bf0 = lds_frag(sV, wn,      32, 0);
    v16h bf1 = lds_frag(sV, wn + 16, 32, 0);

    acc[0][0] = wmma_f16(af0, bf0, acc[0][0]);
    acc[0][1] = wmma_f16(af0, bf1, acc[0][1]);
    acc[1][0] = wmma_f16(af1, bf0, acc[1][0]);
    acc[1][1] = wmma_f16(af1, bf1, acc[1][1]);
    __syncthreads();
  }

  const int crow = (lane >> 4) << 3, ccol = lane & 15;
#pragma unroll
  for (int fm = 0; fm < 2; ++fm)
#pragma unroll
    for (int fn = 0; fn < 2; ++fn)
#pragma unroll
      for (int r = 0; r < 8; ++r) {
        const int mrow = wm + fm * 16 + r + crow;     // 0..127
        const int ncol = wn + fn * 16 + ccol;         // 0..63
        const float val = acc[fm][fn][r] * rin[mrow];
        oh[(size_t)(m0 + mrow) * BINS + h * D + ncol] = (_Float16)val;
      }
}

// ---------------------------------------------------------------------------
// Host launch
// ---------------------------------------------------------------------------

extern "C" void kernel_launch(void* const* d_in, const int* in_sizes, int n_in,
                              void* d_out, int out_size, void* d_ws, size_t ws_size,
                              hipStream_t stream) {
  (void)in_sizes; (void)n_in; (void)out_size; (void)ws_size;

  const float* x    = (const float*)d_in[0];
  const float* prev = (const float*)d_in[1];
  const float* Wq   = (const float*)d_in[2];
  const float* Wk   = (const float*)d_in[3];
  const float* Wv   = (const float*)d_in[4];
  const float* Wo   = (const float*)d_in[5];
  const float* cq   = (const float*)d_in[6];
  const float* ck   = (const float*)d_in[7];
  const float* cv   = (const float*)d_in[8];
  const float* relw = (const float*)d_in[9];

  float* out_final = (float*)d_out;                     // (W, BINS)
  float* qk_out    = out_final + (size_t)Wn * BINS;     // (H, W, W)

  // Workspace carve-out (~53 MB).
  char* p = (char*)d_ws;
  auto carve = [&](size_t bytes) -> void* {
    void* r = (void*)p;
    p += (bytes + 255) & ~(size_t)255;
    return r;
  };
  _Float16* xh  = (_Float16*)carve((size_t)Wn * BINS * 2);
  _Float16* Wqt = (_Float16*)carve((size_t)BINS * BINS * 2);
  _Float16* Wkt = (_Float16*)carve((size_t)BINS * BINS * 2);
  _Float16* Wvt = (_Float16*)carve((size_t)BINS * BINS * 2);
  _Float16* Wot = (_Float16*)carve((size_t)BINS * BINS * 2);
  _Float16* rwT = (_Float16*)carve((size_t)CROP * D * 2);
  float*    xq  = (float*)carve((size_t)Wn * BINS * 4);
  float*    xk  = (float*)carve((size_t)Wn * BINS * 4);
  float*    xv  = (float*)carve((size_t)Wn * BINS * 4);
  _Float16* qh  = (_Float16*)carve((size_t)H * Wn * D * 2);
  _Float16* kh  = (_Float16*)carve((size_t)H * Wn * D * 2);
  _Float16* vT  = (_Float16*)carve((size_t)H * D * Wn * 2);
  _Float16* oh  = (_Float16*)carve((size_t)Wn * BINS * 2);
  float*    rmx = (float*)carve((size_t)H * Wn * 4);
  float*    rin = (float*)carve((size_t)H * Wn * 4);

  const int nWB = BINS * BINS;           // 1M
  const int nXW = Wn * BINS;             // 2M

  // 1) f16 conversions / transposes
  cvt_f32_f16<<<nXW / 256, 256, 0, stream>>>(x, xh, nXW);
  transpose_f32_f16<<<(nWB + 255) / 256, 256, 0, stream>>>(Wq, Wqt, BINS, BINS);
  transpose_f32_f16<<<(nWB + 255) / 256, 256, 0, stream>>>(Wk, Wkt, BINS, BINS);
  transpose_f32_f16<<<(nWB + 255) / 256, 256, 0, stream>>>(Wv, Wvt, BINS, BINS);
  transpose_f32_f16<<<(nWB + 255) / 256, 256, 0, stream>>>(Wo, Wot, BINS, BINS);
  transpose_f32_f16<<<(D * CROP + 255) / 256, 256, 0, stream>>>(relw, rwT, D, CROP);

  // 2) x @ Wq/Wk/Wv  (M=2048, N=1024, K=1024): (2048/128)*(1024/64) = 256 blocks
  gemm_f16<<<256, 256, 0, stream>>>(xh, Wqt, xq, Wn, BINS, BINS);
  gemm_f16<<<256, 256, 0, stream>>>(xh, Wkt, xk, Wn, BINS, BINS);
  gemm_f16<<<256, 256, 0, stream>>>(xh, Wvt, xv, Wn, BINS, BINS);

  // 3) depthwise conv + band split
  conv_bands<<<nXW / 256, 256, 0, stream>>>(xq, xk, xv, cq, ck, cv, qh, kh, vT);

  // 4) qk = (q k^T + bias)/32 + prev  -> output
  qk_kernel<<<dim3(Wn / 64, Wn / 128, H), 256, 0, stream>>>(qh, kh, rwT, prev, qk_out);

  // 5) softmax stats per row
  softmax_stats<<<H * Wn, 256, 0, stream>>>(qk_out, rmx, rin);

  // 6) o = softmax(qk) @ v  (writes (w, bins) layout f16)
  av_kernel<<<dim3(Wn / 128, H), 256, 0, stream>>>(qk_out, vT, rmx, rin, oh);

  // 7) out = o @ Wo
  gemm_f16<<<256, 256, 0, stream>>>(oh, Wot, out_final, Wn, BINS, BINS);
}
